// StableTreeClassifier_15522011808514
// MI455X (gfx1250) — matrix operand order repaired and checked by reading, not compiled, not run on hardware
//
#include <hip/hip_runtime.h>
#include <hip/hip_bf16.h>

// ---------------------------------------------------------------------------
// StableTreeClassifier fused implementation for gfx1250 (MI455X, wave32 WMMA)
//   GEMM1 (x @ Wt)  : v_wmma_f32_16x16x32_bf16, K=256
//   sigmoid + 6-deep soft tree product in LDS
//   GEMM2 (lp @ P)  : v_wmma_f32_16x16x32_bf16, K=64/tree, fp32 accum in regs
//   cross-tree-group reduction via global_atomic_add_f32
// ---------------------------------------------------------------------------

typedef __attribute__((ext_vector_type(16))) __bf16 v16bf;
typedef __attribute__((ext_vector_type(8)))  float  v8f;

#define B_TOTAL  8192
#define F_DIM    256
#define T_TREES  100
#define N_INT    63
#define N_PAD    64
#define C_DIM    100
#define C_PAD    128
#define K2       (T_TREES * N_PAD)   // 6400
#define TPG      5                   // trees per block
#define GROUPS   (T_TREES / TPG)     // 20 tree groups

__device__ __forceinline__ unsigned short f2bf(float f) {
  unsigned int u = __float_as_uint(f);
  u += 0x7FFFu + ((u >> 16) & 1u);           // round-to-nearest-even
  return (unsigned short)(u >> 16);
}

__device__ __forceinline__ float get_temp(const float* lt) {
  return fminf(fmaxf(expf(lt[0]), 0.1f), 5.0f);
}

// ---------------- prep kernels ----------------

__global__ void k_tree_softmax(const float* __restrict__ tw, float* __restrict__ wsoft) {
  __shared__ float red[128];
  const int i = threadIdx.x;
  float v = (i < T_TREES) ? tw[i] : -3.4e38f;
  red[i] = v; __syncthreads();
  for (int s = 64; s > 0; s >>= 1) { if (i < s) red[i] = fmaxf(red[i], red[i + s]); __syncthreads(); }
  const float mx = red[0]; __syncthreads();
  const float e = (i < T_TREES) ? expf(v - mx) : 0.0f;
  red[i] = e; __syncthreads();
  for (int s = 64; s > 0; s >>= 1) { if (i < s) red[i] += red[i + s]; __syncthreads(); }
  if (i < T_TREES) wsoft[i] = e / red[0];
}

__global__ void k_cvt_x(const float* __restrict__ x, unsigned short* __restrict__ xb, int n) {
  const int i = blockIdx.x * blockDim.x + threadIdx.x;
  if (i < n) xb[i] = f2bf(x[i]);
}

// split_weights (T,63,F) fp32 -> wb (T,64,F) bf16, node 63 zero-padded
__global__ void k_cvt_w(const float* __restrict__ sw, unsigned short* __restrict__ wb) {
  const int i = blockIdx.x * blockDim.x + threadIdx.x;
  if (i >= T_TREES * N_PAD * F_DIM) return;
  const int f = i & (F_DIM - 1);
  const int n = (i >> 8) & (N_PAD - 1);
  const int t = i >> 14;
  wb[i] = (n < N_INT) ? f2bf(sw[(t * N_INT + n) * F_DIM + f]) : (unsigned short)0;
}

// wlcpT[c, t*64+l] = wsoft[t] * softmax_c(leaf_logits[t,l,:]/temp), bf16,
// rows c in [100,128) zero.  One block per (t,l).
__global__ void k_leaf_softmax(const float* __restrict__ ll, const float* __restrict__ wsoft,
                               const float* __restrict__ lt, unsigned short* __restrict__ wlcpT) {
  const int tl = blockIdx.x;        // t*64 + l
  const int t  = tl >> 6;
  const int c  = threadIdx.x;       // 0..127
  const float temp = get_temp(lt);
  __shared__ float red[128];
  const float v = (c < C_DIM) ? ll[tl * C_DIM + c] / temp : -3.4e38f;
  red[c] = v; __syncthreads();
  for (int s = 64; s > 0; s >>= 1) { if (c < s) red[c] = fmaxf(red[c], red[c + s]); __syncthreads(); }
  const float mx = red[0]; __syncthreads();
  const float e = (c < C_DIM) ? expf(v - mx) : 0.0f;
  red[c] = e; __syncthreads();
  for (int s = 64; s > 0; s >>= 1) { if (c < s) red[c] += red[c + s]; __syncthreads(); }
  const float p = (c < C_DIM) ? (e / red[0]) * wsoft[t] : 0.0f;
  wlcpT[(size_t)c * K2 + tl] = f2bf(p);
}

__global__ void k_zero(float* __restrict__ out, int n) {
  const int i = blockIdx.x * blockDim.x + threadIdx.x;
  if (i < n) out[i] = 0.0f;
}

// ---------------- fused forest kernel (1 wave / block) ----------------

__global__ void __launch_bounds__(32) k_forest(
    const unsigned short* __restrict__ xb,      // (B,256) bf16
    const unsigned short* __restrict__ wb,      // (T,64,256) bf16
    const float*          __restrict__ split_bias, // (T,63) fp32
    const unsigned short* __restrict__ wlcpT,   // (128,6400) bf16
    const float*          __restrict__ lt,
    float*                __restrict__ out) {   // (B,100) fp32, pre-zeroed
  __shared__ float S [16][N_PAD];   // split probs for current tree
  __shared__ float L2[16][N_PAD];   // leaf probs for current tree

  const int lane  = threadIdx.x;
  const int nco   = lane & 15;      // column within a 16-wide WMMA tile
  const int kh    = lane >> 4;      // half-wave selector
  const int b0    = blockIdx.x * 16;
  const int tbase = blockIdx.y * TPG;
  const float invt = 1.0f / get_temp(lt);

  const v8f vzero = {0.f, 0.f, 0.f, 0.f, 0.f, 0.f, 0.f, 0.f};
  v8f outacc[8];
  #pragma unroll
  for (int j = 0; j < 8; ++j) outacc[j] = vzero;

  const int arow = b0 + nco;        // A-matrix row (batch index)

  for (int it = 0; it < TPG; ++it) {
    const int t = tbase + it;

    // ---- GEMM1: 16(b) x 64(n) logits tile, K = 256 ----
    v8f acc1[4];
    #pragma unroll
    for (int j = 0; j < 4; ++j) acc1[j] = vzero;

    #pragma unroll 2
    for (int k = 0; k < F_DIM; k += 32) {
      union { v16bf v; uint4 q[2]; } a;
      const unsigned short* ap = xb + (size_t)arow * F_DIM + k + kh * 8;
      a.q[0] = *(const uint4*)(ap);          // K chunk [kh*8 .. kh*8+7]
      a.q[1] = *(const uint4*)(ap + 16);     // K chunk [16+kh*8 .. 16+kh*8+7]
      #pragma unroll
      for (int jn = 0; jn < 4; ++jn) {
        union { v16bf v; uint4 q[2]; } b;
        const unsigned short* bp =
            wb + ((size_t)(t * N_PAD + jn * 16 + nco) * F_DIM + k + kh * 8);
        b.q[0] = *(const uint4*)(bp);
        b.q[1] = *(const uint4*)(bp + 16);
        acc1[jn] = __builtin_amdgcn_wmma_f32_16x16x32_bf16(
            false, a.v, false, b.v, (short)0, acc1[jn], false, false);
      }
    }

    // ---- bias + sigmoid(logit/temp) -> S ----
    #pragma unroll
    for (int jn = 0; jn < 4; ++jn) {
      const int n = jn * 16 + nco;
      const float bias = (n < N_INT) ? split_bias[t * N_INT + n] : 0.0f;
      #pragma unroll
      for (int r = 0; r < 8; ++r) {
        const float z = (acc1[jn][r] + bias) * invt;
        S[r + 8 * kh][n] = 1.0f / (1.0f + __expf(-z));
      }
    }
    __syncthreads();

    // ---- 6-deep soft decision path products: 16 rows x 64 leaves ----
    #pragma unroll 4
    for (int q = 0; q < 32; ++q) {
      const int p    = q * 32 + lane;
      const int row  = p >> 6;
      const int leaf = p & 63;
      float prob = 1.0f;
      int node = leaf + N_INT;
      #pragma unroll
      for (int d = 0; d < 6; ++d) {
        const int parent = (node - 1) >> 1;
        const float sp = S[row][parent];
        prob *= (node & 1) ? (1.0f - sp) : sp;   // even child = right = sp
        node = parent;
      }
      L2[row][leaf] = prob;
    }
    __syncthreads();

    // ---- GEMM2: out(16 x 128) += L2(16x64) x wlcpT'(64x128), K = 64 ----
    #pragma unroll
    for (int ks = 0; ks < 2; ++ks) {
      union { v16bf v; unsigned short u[16]; } a2;
      #pragma unroll
      for (int e = 0; e < 8; ++e)
        a2.u[e]     = f2bf(L2[nco][ks * 32 + kh * 8 + e]);
      #pragma unroll
      for (int e = 0; e < 8; ++e)
        a2.u[8 + e] = f2bf(L2[nco][ks * 32 + 16 + kh * 8 + e]);
      #pragma unroll
      for (int jc = 0; jc < 8; ++jc) {
        union { v16bf v; uint4 q[2]; } b2;
        const int c = jc * 16 + nco;
        const unsigned short* bp =
            wlcpT + (size_t)c * K2 + t * N_PAD + ks * 32 + kh * 8;
        b2.q[0] = *(const uint4*)(bp);
        b2.q[1] = *(const uint4*)(bp + 16);
        outacc[jc] = __builtin_amdgcn_wmma_f32_16x16x32_bf16(
            false, a2.v, false, b2.v, (short)0, outacc[jc], false, false);
      }
    }
    __syncthreads();
  }

  // ---- accumulate tree-group partials into out ----
  #pragma unroll
  for (int jc = 0; jc < 8; ++jc) {
    const int c = jc * 16 + nco;
    if (c < C_DIM) {
      #pragma unroll
      for (int r = 0; r < 8; ++r) {
        const int row = b0 + r + 8 * kh;
        atomicAdd(out + (size_t)row * C_DIM + c, outacc[jc][r]);
      }
    }
  }
}

// ---------------- launcher ----------------

extern "C" void kernel_launch(void* const* d_in, const int* in_sizes, int n_in,
                              void* d_out, int out_size, void* d_ws, size_t ws_size,
                              hipStream_t stream) {
  const float* x  = (const float*)d_in[0];   // (8192,256)
  const float* sw = (const float*)d_in[1];   // (100,63,256)
  const float* sb = (const float*)d_in[2];   // (100,63)
  const float* ll = (const float*)d_in[3];   // (100,64,100)
  const float* tw = (const float*)d_in[4];   // (100,)
  const float* lt = (const float*)d_in[5];   // scalar
  float* out = (float*)d_out;                // (8192,100)

  char* ws = (char*)d_ws;
  float*          wsoft = (float*)ws;                                   //   512 B
  unsigned short* xb    = (unsigned short*)(ws + 512);                  // 4.19 MB
  unsigned short* wb    = (unsigned short*)(ws + 512 + 4194304);        // 3.28 MB
  unsigned short* wlcpT = (unsigned short*)(ws + 512 + 4194304 + 3276800); // 1.64 MB

  k_tree_softmax<<<1, 128, 0, stream>>>(tw, wsoft);
  k_cvt_x<<<(B_TOTAL * F_DIM + 255) / 256, 256, 0, stream>>>(x, xb, B_TOTAL * F_DIM);
  k_cvt_w<<<(T_TREES * N_PAD * F_DIM + 255) / 256, 256, 0, stream>>>(sw, wb);
  k_leaf_softmax<<<T_TREES * N_PAD, 128, 0, stream>>>(ll, wsoft, lt, wlcpT);
  k_zero<<<(B_TOTAL * C_DIM + 255) / 256, 256, 0, stream>>>(out, B_TOTAL * C_DIM);

  dim3 grid(B_TOTAL / 16, GROUPS);
  k_forest<<<grid, 32, 0, stream>>>(xb, wb, sb, wlcpT, lt, out);
}
